// SpikingNet_26688926777687
// MI455X (gfx1250) — matrix operand order repaired and checked by reading, not compiled, not run on hardware
//
#include <hip/hip_runtime.h>
#include <hip/hip_bf16.h>

typedef __bf16 v8bf  __attribute__((ext_vector_type(8)));
typedef __bf16 v16bf __attribute__((ext_vector_type(16)));
typedef float  v8f   __attribute__((ext_vector_type(8)));

#define T_STEPS 25
#define BATCH   4096
#define D_IN    784
#define D_H     1000
#define D_OUT   10
#define M_ROWS  (T_STEPS * BATCH)   /* 102400 */
#define K1P     800                 /* D_IN padded to multiple of 32 */
#define N1P     1024                /* D_H padded to multiple of 16 */
#define K2P     1024                /* D_H padded (GEMM2 K) */
#define N2P     16                  /* D_OUT padded */
#define BETA    0.95f

// fp32 -> bf16 round-to-nearest-even, as raw bits
__device__ __forceinline__ unsigned short f2bf_bits(float f) {
    unsigned int u = __float_as_uint(f);
    u += 0x7fffu + ((u >> 16) & 1u);
    return (unsigned short)(u >> 16);
}

__device__ __forceinline__ v16bf cat8(v8bf lo, v8bf hi) {
    return __builtin_shufflevector(lo, hi, 0,1,2,3,4,5,6,7,8,9,10,11,12,13,14,15);
}

// ---------- conversion kernels (fp32 -> bf16, zero-padded) ----------
__global__ void conv_x_kernel(const float* __restrict__ x, unsigned short* __restrict__ xb) {
    size_t i = (size_t)blockIdx.x * blockDim.x + threadIdx.x;
    if (i >= (size_t)M_ROWS * K1P) return;
    int    k = (int)(i % K1P);
    size_t m = i / K1P;
    float v = (k < D_IN) ? x[m * D_IN + k] : 0.0f;
    xb[i] = f2bf_bits(v);
}

__global__ void conv_w1_kernel(const float* __restrict__ w1, unsigned short* __restrict__ w1b) {
    size_t i = (size_t)blockIdx.x * blockDim.x + threadIdx.x;
    if (i >= (size_t)N1P * K1P) return;
    int k = (int)(i % K1P);
    int n = (int)(i / K1P);
    float v = (n < D_H && k < D_IN) ? w1[(size_t)n * D_IN + k] : 0.0f;
    w1b[i] = f2bf_bits(v);
}

__global__ void conv_w2_kernel(const float* __restrict__ w2, unsigned short* __restrict__ w2b) {
    size_t i = (size_t)blockIdx.x * blockDim.x + threadIdx.x;
    if (i >= (size_t)N2P * K2P) return;
    int k = (int)(i % K2P);
    int n = (int)(i / K2P);
    float v = (n < D_OUT && k < D_H) ? w2[(size_t)n * D_H + k] : 0.0f;
    w2b[i] = f2bf_bits(v);
}

// ---------- WMMA GEMM:  C(M,ldc) = A(M,K) * B(N,K)^T + bias ----------
// Block = 8 waves covering 128 rows (one 16-row M-tile per wave) x NCOLS cols.
// The NCOLS x K strip of B is staged ONCE per block into LDS using
// global_load_async_to_lds_b128 (ASYNCcnt), then read back via ds_load_b128.
// Inner loop is software-pipelined: A fragment (global) double-buffered one
// k-step ahead, B fragment (LDS) prefetched one WMMA ahead, so the waits
// in front of each v_wmma only cover loads issued a full stage earlier.
template <int NCOLS>
__global__ void gemm_wmma_bf16(const unsigned short* __restrict__ A,
                               const unsigned short* __restrict__ Bw,
                               const float* __restrict__ bias, int nbias,
                               float* __restrict__ C,
                               int K, int N, int ldc) {
    constexpr int JT = NCOLS / 16;
    extern __shared__ unsigned short Bs[];      // NCOLS * (K+8) bf16, row-padded
    const int LK = K + 8;                       // LDS row stride (elements)

    int stripsPerRow = N / NCOLS;
    int bm = blockIdx.x / stripsPerRow;         // group of 8 M-tiles (128 rows)
    int bn = blockIdx.x % stripsPerRow;         // NCOLS-wide column strip
    int tid  = threadIdx.x;
    int wv   = tid >> 5;
    int lane = tid & 31;
    int m    = lane & 15;                       // row in A tile / col in C tile
    int half = lane >> 4;

    // ---- stage B strip -> LDS (async, 16B chunks) ----
    {
        const unsigned short* gsrc = Bw + (size_t)bn * NCOLS * K;  // strip = contiguous rows
        const int CPR = (K * 2) / 16;                              // 16B chunks per row
        const int totalChunks = NCOLS * CPR;
        unsigned ldsBase = (unsigned)(uintptr_t)(&Bs[0]);          // low 32 bits = LDS offset
        for (int q = tid; q < totalChunks; q += 256) {
            int r     = q / CPR;
            int cbyte = (q - r * CPR) * 16;
            unsigned voff  = (unsigned)(r * K * 2 + cbyte);
            unsigned laddr = ldsBase + (unsigned)(r * LK * 2 + cbyte);
            asm volatile("global_load_async_to_lds_b128 %0, %1, %2"
                         :: "v"(laddr), "v"(voff), "s"(gsrc) : "memory");
        }
        asm volatile("s_wait_asynccnt 0x0" ::: "memory");
        __syncthreads();
    }

    int mt = bm * 8 + wv;                       // this wave's M-tile

    v8f zero = {0.f, 0.f, 0.f, 0.f, 0.f, 0.f, 0.f, 0.f};
    v8f acc[JT];
#pragma unroll
    for (int j = 0; j < JT; ++j) acc[j] = zero;

    const __bf16* arow = (const __bf16*)A + (size_t)(mt * 16 + m) * K;
    const __bf16* Bsb  = (const __bf16*)Bs;

    // B 32x16 fragment from LDS: lane column = m, elems 0-7 -> K=kk+half*8.., elems 8-15 -> K=kk+16+half*8..
    auto loadB = [&](int kk, int j) -> v16bf {
        const __bf16* brow = Bsb + (size_t)(j * 16 + m) * LK + kk + half * 8;
        return cat8(*(const v8bf*)brow, *(const v8bf*)(brow + 16));
    };

    // ---- software-pipelined main loop ----
    // A 16x32 bf16 fragment: elems 0-7 -> K=k+half*8.., elems 8-15 -> K=k+16+half*8..
    v8bf alo = *(const v8bf*)(arow + half * 8);
    v8bf ahi = *(const v8bf*)(arow + 16 + half * 8);
    v16bf bcur = loadB(0, 0);

    for (int k = 0; k < K; k += 32) {
        v16bf afrag = cat8(alo, ahi);
        int kn = (k + 32 < K) ? (k + 32) : k;          // clamped prefetch (no branch)
        alo = *(const v8bf*)(arow + kn + half * 8);
        ahi = *(const v8bf*)(arow + kn + 16 + half * 8);
#pragma unroll
        for (int j = 0; j < JT; ++j) {
            int jn = (j + 1 == JT) ? 0 : (j + 1);
            int kb = (j + 1 == JT) ? kn : k;
            v16bf bnext = loadB(kb, jn);               // issue before consuming bcur
            acc[j] = __builtin_amdgcn_wmma_f32_16x16x32_bf16(
                false, afrag, false, bcur, (short)0, acc[j], false, false);
            bcur = bnext;
        }
    }

    // C/D layout: VGPR v holds row (mt*16 + v + 8*half), column (lane&15) of tile
#pragma unroll
    for (int j = 0; j < JT; ++j) {
        int ncol = bn * NCOLS + j * 16 + m;
        float bv = (ncol < nbias) ? bias[ncol] : 0.0f;
        float* cbase = C + (size_t)(mt * 16 + half * 8) * ldc + ncol;
#pragma unroll
        for (int v = 0; v < 8; ++v)
            cbase[(size_t)v * ldc] = acc[j][v] + bv;
    }
}

// ---------- layer-1 membrane scan: parallel over (b,h), sequential over t ----------
__global__ void scan1_kernel(const float* __restrict__ C1, unsigned short* __restrict__ spk1) {
    int idx = blockIdx.x * blockDim.x + threadIdx.x;   // BATCH * N1P threads
    int b = idx >> 10;
    int h = idx & (N1P - 1);
    if (h >= D_H) {  // zero the K-padding columns of spk1 for GEMM2
        for (int t = 0; t < T_STEPS; ++t)
            spk1[((size_t)(t * BATCH + b) << 10) + h] = 0;
        return;
    }
    const unsigned short ONE_BF16 = 0x3F80;
    float mem = 0.0f;
    for (int t = 0; t < T_STEPS; ++t) {
        size_t off = ((size_t)(t * BATCH + b) << 10) + h;
        float cur   = C1[off];
        float reset = (mem > 1.0f) ? 1.0f : 0.0f;   // spike(mem - THR), prior mem
        mem = BETA * mem + cur - reset;
        spk1[off] = (mem > 1.0f) ? ONE_BF16 : (unsigned short)0;
    }
}

// ---------- layer-2 membrane scan + output ----------
__global__ void scan2_kernel(const float* __restrict__ cur2, float* __restrict__ out) {
    int idx = blockIdx.x * blockDim.x + threadIdx.x;   // BATCH * D_OUT threads
    if (idx >= BATCH * D_OUT) return;
    int b = idx / D_OUT;
    int n = idx - b * D_OUT;
    float mem = 0.0f;
    for (int t = 0; t < T_STEPS; ++t) {
        float cur   = cur2[((size_t)(t * BATCH + b) << 4) + n];
        float reset = (mem > 1.0f) ? 1.0f : 0.0f;
        mem = BETA * mem + cur - reset;
        out[(size_t)t * BATCH * D_OUT + idx] = (mem > 1.0f) ? 1.0f : 0.0f;  // spk_rec
    }
    out[(size_t)T_STEPS * BATCH * D_OUT + idx] = mem;                        // mem2
}

extern "C" void kernel_launch(void* const* d_in, const int* in_sizes, int n_in,
                              void* d_out, int out_size, void* d_ws, size_t ws_size,
                              hipStream_t stream) {
    const float* x  = (const float*)d_in[0];   // (25, 4096, 784)
    const float* w1 = (const float*)d_in[1];   // (1000, 784)
    const float* b1 = (const float*)d_in[2];   // (1000,)
    const float* w2 = (const float*)d_in[3];   // (10, 1000)
    const float* b2 = (const float*)d_in[4];   // (10,)
    float* out = (float*)d_out;

    // workspace layout (all region sizes are multiples of 256 bytes)
    char* ws = (char*)d_ws;
    size_t off = 0;
    unsigned short* xb   = (unsigned short*)(ws + off); off += (size_t)M_ROWS * K1P * 2;  // 163.8 MB
    unsigned short* w1b  = (unsigned short*)(ws + off); off += (size_t)N1P * K1P * 2;     //   1.6 MB
    float*          C1   = (float*)(ws + off);          off += (size_t)M_ROWS * N1P * 4;  // 419.4 MB
    unsigned short* spk1 = (unsigned short*)(ws + off); off += (size_t)M_ROWS * K2P * 2;  // 209.7 MB
    unsigned short* w2b  = (unsigned short*)(ws + off); off += (size_t)N2P * K2P * 2;     //  32 KB
    float*          cur2 = (float*)(ws + off);          off += (size_t)M_ROWS * N2P * 4;  //   6.6 MB

    // 1) convert inputs to zero-padded bf16
    conv_x_kernel <<<(int)(((size_t)M_ROWS * K1P) / 256), 256, 0, stream>>>(x, xb);
    conv_w1_kernel<<<(int)(((size_t)N1P * K1P) / 256),    256, 0, stream>>>(w1, w1b);
    conv_w2_kernel<<<(int)(((size_t)N2P * K2P) / 256),    256, 0, stream>>>(w2, w2b);

    // 2) GEMM1: C1(102400,1024) = xb(102400,800) * w1b(1024,800)^T + b1
    //    blocks = (6400/8 M-groups) * (1024/64 strips) = 12800; LDS = 64*(800+8)*2 B
    gemm_wmma_bf16<64><<<12800, 256, 64 * (K1P + 8) * 2, stream>>>(
        xb, w1b, b1, D_H, C1, K1P, N1P, N1P);

    // 3) membrane scan layer 1 -> spk1 (bf16, K-padded to 1024)
    scan1_kernel<<<(BATCH * N1P) / 256, 256, 0, stream>>>(C1, spk1);

    // 4) GEMM2: cur2(102400,16) = spk1(102400,1024) * w2b(16,1024)^T + b2
    //    blocks = 6400/8 = 800; LDS = 16*(1024+8)*2 B
    gemm_wmma_bf16<16><<<800, 256, 16 * (K2P + 8) * 2, stream>>>(
        spk1, w2b, b2, D_OUT, cur2, K2P, N2P, N2P);

    // 5) membrane scan layer 2 -> spk_rec ++ mem2 in d_out
    scan2_kernel<<<(BATCH * D_OUT) / 256, 256, 0, stream>>>(cur2, out);
}
